// FEDformer1D_22651657519306
// MI455X (gfx1250) — matrix operand order repaired and checked by reading, not compiled, not run on hardware
//
#include <hip/hip_runtime.h>
#include <hip/hip_bf16.h>

// FourierCrossAttention (FEDformer) on MI455X / gfx1250.
// All heavy math lowered to v_wmma_f32_16x16x32_bf16 (fp32 accumulate).
// LDS tiles are stored in *fragment order* (per-lane 16 contiguous bf16), so
// each WMMA operand load is two ds_load_b128 instead of 16 ds_load_u16.

typedef __attribute__((ext_vector_type(16))) __bf16 v16bf;
typedef __attribute__((ext_vector_type(8)))  __bf16 v8bf;
typedef __attribute__((ext_vector_type(8)))  float  v8f;

#define BATCH 32
#define LSEQ  1024
#define NH    8
#define NE    64
#define NM    64
#define NO    64

__device__ __forceinline__ v8f zero8() {
  v8f z = {0.f,0.f,0.f,0.f,0.f,0.f,0.f,0.f};
  return z;
}

// ---- swizzled tile element offsets (fragment order, ISA 7.12.2 wave32) ----
// A tile 16x32: lane = (m&15) + 16*((k>>3)&1); lane data = k runs {0..7|8..15} and {16..23|24..31}
__device__ __forceinline__ int swzA(int m, int k) {
  return (((m & 15) + (((k >> 3) & 1) << 4)) << 4) + ((k >> 4) << 3) + (k & 7);
}
// B tile 32x16: lane = (n&15) + 16*((k>>3)&1); lane data = k runs {0..7|8..15} and {16..23|24..31}
__device__ __forceinline__ int swzB(int k, int n) {
  return (((n & 15) + (((k >> 3) & 1) << 4)) << 4) + ((k >> 4) << 3) + (k & 7);
}
// fragment load: two 16B LDS reads per lane
__device__ __forceinline__ v16bf load_frag(const __bf16* tile) {
  const int lane = threadIdx.x & 31;
  const v8bf* p = (const v8bf*)(tile + lane * 16);
  v8bf lo = p[0];
  v8bf hi = p[1];
  return __builtin_shufflevector(lo, hi, 0,1,2,3,4,5,6,7,8,9,10,11,12,13,14,15);
}
__device__ __forceinline__ v8f wmma_bf16(v16bf a, v16bf b, v8f c) {
  return __builtin_amdgcn_wmma_f32_16x16x32_bf16(false, a, false, b, (short)0, c, false, false);
}

// ---------------- K0: DFT / iDFT bases ----------------
// basisF [c=128][t=1024] : c<64 -> cos(2*pi*m*t/L), c>=64 -> -sin(...)
// basisI [t=1024][c=128] : folded irfft + 1/(512*512) scaling
__global__ __launch_bounds__(256) void k_basis(__bf16* basisF, __bf16* basisI) {
  const float w0 = 6.28318530717958647692f / 1024.f;
  const float s0 = 1.f / (1024.f * 262144.f);
  for (int idx = blockIdx.x*blockDim.x + threadIdx.x; idx < 128*LSEQ;
       idx += gridDim.x*blockDim.x) {
    int t = idx & 1023, c = idx >> 10, m = c & 63;
    float th = (float)((m * t) & 1023) * w0;
    basisF[idx] = (__bf16)((c < 64) ? __cosf(th) : -__sinf(th));
  }
  for (int idx = blockIdx.x*blockDim.x + threadIdx.x; idx < 128*LSEQ;
       idx += gridDim.x*blockDim.x) {
    int c = idx & 127, t = idx >> 7, m = c & 63;
    float th = (float)((m * t) & 1023) * w0;
    float vi;
    if (c < 64) vi = ((m == 0) ? 1.f : 2.f) * s0 * __cosf(th);
    else        vi = ((m == 0) ? 0.f : -2.f * s0 * __sinf(th));
    basisI[idx] = (__bf16)vi;
  }
}

// ---------------- K0b: repack W -> bf16 [ri][h][x][o][e] ----------------
__global__ __launch_bounds__(256) void k_prepw(const float* Wr, const float* Wi, __bf16* wb) {
  for (long idx = blockIdx.x*blockDim.x + threadIdx.x; idx < 2L*NH*64*NO*NE;
       idx += (long)gridDim.x*blockDim.x) {
    int e  = idx & 63;
    int o  = (idx >> 6) & 63;
    int xm = (idx >> 12) & 63;
    int h  = (idx >> 18) & 7;
    int ri = (int)(idx >> 21) & 1;
    const float* src = ri ? Wi : Wr;
    wb[idx] = (__bf16)src[(((size_t)h*NE + e)*NO + o)*NM + xm];
  }
}

// ---------------- K1: truncated DFT of q and k ----------------
// grid 512: [src(2)][b(32)][h(8)]
// out: qft [bh][ri][x][e], kf [bh][ri][e][y], kft [bh][ri][y][e]
__global__ __launch_bounds__(256) void k_dft(const float* q, const float* k,
                                             const __bf16* basisF,
                                             __bf16* qft, __bf16* kf, __bf16* kft) {
  __shared__ __align__(16) __bf16 sA[16*512];   // tiles [ks*4+mt]
  __shared__ __align__(16) __bf16 sB[32*512];   // tiles [ks*8+nt]
  const int bid = blockIdx.x;
  const int src = bid >> 8;
  const int bh  = bid & 255;
  const int b = bh >> 3, h = bh & 7;
  const int tid = threadIdx.x;
  const int w = tid >> 5, lane = tid & 31;
  const int mt = w >> 1, ntb = (w & 1) * 4;
  const float* xb = (src ? k : q) + ((size_t)b*LSEQ*NH + h) * (size_t)NE;

  v8f acc[4];
#pragma unroll
  for (int j = 0; j < 4; ++j) acc[j] = zero8();

  for (int tc = 0; tc < 8; ++tc) {
    const int t0 = tc * 128;
    // stage x chunk into A-swizzled tiles (transpose; element-wise, coalesced reads)
#pragma unroll 4
    for (int i = tid; i < 64*128; i += 256) {
      int t = i >> 6, e = i & 63;
      float v = xb[(size_t)(t0 + t)*(NH*NE) + e];
      sA[(((t >> 5) << 2) + (e >> 4))*512 + swzA(e & 15, t & 31)] = (__bf16)v;
    }
    // stage basis chunk into B-swizzled tiles (vector 16B copies)
#pragma unroll 4
    for (int i = tid; i < 128*16; i += 256) {
      int c = i >> 4, tl = (i & 15) << 3;
      v8bf val = *(const v8bf*)(basisF + (size_t)c*1024 + t0 + tl);
      *(v8bf*)(&sB[(((tl >> 5) << 3) + (c >> 4))*512 + swzB(tl & 31, c & 15)]) = val;
    }
    __syncthreads();
#pragma unroll
    for (int ks = 0; ks < 4; ++ks) {
      v16bf a = load_frag(&sA[((ks << 2) + mt)*512]);
#pragma unroll
      for (int j = 0; j < 4; ++j) {
        v16bf bb = load_frag(&sB[((ks << 3) + ntb + j)*512]);
        acc[j] = wmma_bf16(a, bb, acc[j]);
      }
    }
    __syncthreads();
  }
  // store: D[e][c] ; c<64 real modes, c>=64 imag modes
  const int hi = lane >> 4, n = lane & 15;
#pragma unroll
  for (int j = 0; j < 4; ++j) {
    int c = (ntb + j)*16 + n, ri = c >> 6, mc = c & 63;
#pragma unroll
    for (int v = 0; v < 8; ++v) {
      int e = mt*16 + v + 8*hi;
      __bf16 val = (__bf16)acc[j][v];
      if (src == 0) {
        qft[(((size_t)bh*2 + ri)*64 + mc)*64 + e] = val;  // [x][e]
      } else {
        kf [(((size_t)bh*2 + ri)*64 + e )*64 + mc] = val; // [e][y]
        kft[(((size_t)bh*2 + ri)*64 + mc)*64 + e ] = val; // [y][e]
      }
    }
  }
}

// ---------------- K2: S = Qf^T Kf ; T = ctanh(S) ; U = Kf T^T ----------------
// grid 256 (b,h); out ux [h][x][ri][b][e]
__global__ __launch_bounds__(256) void k_attn(const __bf16* qft, const __bf16* kf,
                                              const __bf16* kft, __bf16* ux) {
  __shared__ __align__(16) __bf16 sQrA[8*512], sQiA[8*512]; // A tiles [ks*4+mt]; reused as T^T B tiles
  __shared__ __align__(16) __bf16 sKrB[8*512], sKiB[8*512]; // B tiles [ks*4+nt]
  __shared__ __align__(16) __bf16 sKrA[8*512], sKiA[8*512]; // A tiles [ks*4+mt]
  const int bh = blockIdx.x;
  const int b = bh >> 3, h = bh & 7;
  const int tid = threadIdx.x;
  const int w = tid >> 5, lane = tid & 31;
  const int mt = w >> 1, ntb = (w & 1) * 2;
  const int hi = lane >> 4, n = lane & 15;

  const __bf16* qb  = qft + (size_t)bh*2*64*64;
  const __bf16* kb  = kf  + (size_t)bh*2*64*64;
  const __bf16* ktb = kft + (size_t)bh*2*64*64;

  // ---- vector staging (16B global load -> 16B LDS store) ----
#pragma unroll 2
  for (int i = tid; i < 2*64*8; i += 256) {   // Q as A: m=x, k=e
    int ri = i >> 9, x = (i >> 3) & 63, e0 = (i & 7) << 3;
    v8bf val = *(const v8bf*)(qb + ((size_t)ri*64 + x)*64 + e0);
    __bf16* dst = ri ? sQiA : sQrA;
    *(v8bf*)(dst + (((e0 >> 5) << 2) + (x >> 4))*512 + swzA(x & 15, e0 & 31)) = val;
  }
#pragma unroll 2
  for (int i = tid; i < 2*64*8; i += 256) {   // K as B: k=e, n=y  (from kft [y][e])
    int ri = i >> 9, y = (i >> 3) & 63, e0 = (i & 7) << 3;
    v8bf val = *(const v8bf*)(ktb + ((size_t)ri*64 + y)*64 + e0);
    __bf16* dst = ri ? sKiB : sKrB;
    *(v8bf*)(dst + (((e0 >> 5) << 2) + (y >> 4))*512 + swzB(e0 & 31, y & 15)) = val;
  }
#pragma unroll 2
  for (int i = tid; i < 2*64*8; i += 256) {   // K as A: m=e, k=y  (from kf [e][y])
    int ri = i >> 9, e = (i >> 3) & 63, y0 = (i & 7) << 3;
    v8bf val = *(const v8bf*)(kb + ((size_t)ri*64 + e)*64 + y0);
    __bf16* dst = ri ? sKiA : sKrA;
    *(v8bf*)(dst + (((y0 >> 5) << 2) + (e >> 4))*512 + swzA(e & 15, y0 & 31)) = val;
  }
  __syncthreads();

  // ---- S = Qf^T Kf (complex); separate accumulators, no NEG modifier ----
  v8f aA[2], aB[2], aC[2], aD[2];
#pragma unroll
  for (int p = 0; p < 2; ++p) { aA[p]=zero8(); aB[p]=zero8(); aC[p]=zero8(); aD[p]=zero8(); }
#pragma unroll
  for (int ks = 0; ks < 2; ++ks) {
    v16bf aQr = load_frag(&sQrA[((ks << 2) + mt)*512]);
    v16bf aQi = load_frag(&sQiA[((ks << 2) + mt)*512]);
#pragma unroll
    for (int p = 0; p < 2; ++p) {
      int nt = ntb + p;
      v16bf bKr = load_frag(&sKrB[((ks << 2) + nt)*512]);
      v16bf bKi = load_frag(&sKiB[((ks << 2) + nt)*512]);
      aA[p] = wmma_bf16(aQr, bKr, aA[p]);
      aB[p] = wmma_bf16(aQi, bKi, aB[p]);
      aC[p] = wmma_bf16(aQr, bKi, aC[p]);
      aD[p] = wmma_bf16(aQi, bKr, aD[p]);
    }
  }
  __syncthreads();   // all waves done reading Q tiles before reuse as T^T

  // ---- complex tanh in registers, write T^T (B-swizzled) into reused Q LDS ----
  __bf16* sTrB = sQrA;
  __bf16* sTiB = sQiA;
#pragma unroll
  for (int p = 0; p < 2; ++p) {
    v8f sr = aA[p] - aB[p];
    v8f si = aC[p] + aD[p];
    int nt = ntb + p;
#pragma unroll
    for (int v = 0; v < 8; ++v) {
      float a2 = 2.f*sr[v], b2 = 2.f*si[v];
      float sech = 1.f / coshf(a2);                 // 0 on overflow -> correct limit
      float den  = 1.f + __cosf(b2) * sech;
      float re = tanhf(a2) / den;
      float im = __sinf(b2) * sech / den;
      int xg = mt*16 + v + 8*hi;                    // row of S (mode x)
      int y  = nt*16 + n;                           // col of S (mode y)
      int off = (((y >> 5) << 2) + (xg >> 4))*512 + swzB(y & 31, xg & 15);
      sTrB[off] = (__bf16)re;
      sTiB[off] = (__bf16)im;
    }
  }
  __syncthreads();

  // ---- U = Kf @ T^T (complex) ----
#pragma unroll
  for (int p = 0; p < 2; ++p) { aA[p]=zero8(); aB[p]=zero8(); aC[p]=zero8(); aD[p]=zero8(); }
#pragma unroll
  for (int ks = 0; ks < 2; ++ks) {
    v16bf aKr = load_frag(&sKrA[((ks << 2) + mt)*512]);
    v16bf aKi = load_frag(&sKiA[((ks << 2) + mt)*512]);
#pragma unroll
    for (int p = 0; p < 2; ++p) {
      int nt = ntb + p;
      v16bf bTr = load_frag(&sTrB[((ks << 2) + nt)*512]);
      v16bf bTi = load_frag(&sTiB[((ks << 2) + nt)*512]);
      aA[p] = wmma_bf16(aKr, bTr, aA[p]);
      aB[p] = wmma_bf16(aKi, bTi, aB[p]);
      aC[p] = wmma_bf16(aKr, bTi, aC[p]);
      aD[p] = wmma_bf16(aKi, bTr, aD[p]);
    }
  }
#pragma unroll
  for (int p = 0; p < 2; ++p) {
    v8f ur = aA[p] - aB[p];
    v8f ui = aC[p] + aD[p];
    int nt = ntb + p;
#pragma unroll
    for (int v = 0; v < 8; ++v) {
      int e = mt*16 + v + 8*hi;
      int x = nt*16 + n;
      size_t base = (((size_t)h*64 + x)*2) * (BATCH*NE);
      ux[base +                    (size_t)b*NE + e] = (__bf16)ur[v];
      ux[base + (size_t)BATCH*NE + (size_t)b*NE + e] = (__bf16)ui[v];
    }
  }
}

// ---------------- K3: G[b,o] = sum_e U[b,e] * W[e,o]  (complex), per (h,x) ----------------
// grid 512 (h*64+x), 128 threads; out gc [b][h][o][c], c = ri*64 + x
__global__ __launch_bounds__(128) void k_wcon(const __bf16* ux, const __bf16* wb,
                                              __bf16* gc) {
  __shared__ __align__(16) __bf16 sUrA[4*512], sUiA[4*512]; // A tiles [ks*2+mt]
  __shared__ __align__(16) __bf16 sWrB[8*512], sWiB[8*512]; // B tiles [ks*4+nt]
  const int h  = blockIdx.x >> 6;
  const int xm = blockIdx.x & 63;
  const int tid = threadIdx.x;
  const int w = tid >> 5, lane = tid & 31;
  const int mt = w & 1, ntb = (w >> 1) * 2;
  const int hi = lane >> 4, n = lane & 15;

  const __bf16* ub = ux + (((size_t)h*64 + xm)*2) * (BATCH*NE);
  const __bf16* wr = wb + (((size_t)0*NH + h)*64 + xm) * (size_t)(NO*NE);
  const __bf16* wi = wb + (((size_t)1*NH + h)*64 + xm) * (size_t)(NO*NE);
#pragma unroll 2
  for (int i = tid; i < 2*32*8; i += 128) {   // U as A: m=b, k=e
    int ri = i >> 8, b = (i >> 3) & 31, e0 = (i & 7) << 3;
    v8bf val = *(const v8bf*)(ub + ((size_t)ri*32 + b)*64 + e0);
    __bf16* dst = ri ? sUiA : sUrA;
    *(v8bf*)(dst + (((e0 >> 5) << 1) + (b >> 4))*512 + swzA(b & 15, e0 & 31)) = val;
  }
#pragma unroll 2
  for (int i = tid; i < 2*64*8; i += 128) {   // W as B: k=e, n=o
    int ri = i >> 9, o = (i >> 3) & 63, e0 = (i & 7) << 3;
    v8bf val = *(const v8bf*)((ri ? wi : wr) + (size_t)o*64 + e0);
    __bf16* dst = ri ? sWiB : sWrB;
    *(v8bf*)(dst + (((e0 >> 5) << 2) + (o >> 4))*512 + swzB(e0 & 31, o & 15)) = val;
  }
  __syncthreads();

  v8f aA[2], aB[2], aC[2], aD[2];
#pragma unroll
  for (int p = 0; p < 2; ++p) { aA[p]=zero8(); aB[p]=zero8(); aC[p]=zero8(); aD[p]=zero8(); }
#pragma unroll
  for (int ks = 0; ks < 2; ++ks) {
    v16bf aUr = load_frag(&sUrA[((ks << 1) + mt)*512]);
    v16bf aUi = load_frag(&sUiA[((ks << 1) + mt)*512]);
#pragma unroll
    for (int p = 0; p < 2; ++p) {
      int nt = ntb + p;
      v16bf bWr = load_frag(&sWrB[((ks << 2) + nt)*512]);
      v16bf bWi = load_frag(&sWiB[((ks << 2) + nt)*512]);
      aA[p] = wmma_bf16(aUr, bWr, aA[p]);
      aB[p] = wmma_bf16(aUi, bWi, aB[p]);
      aC[p] = wmma_bf16(aUr, bWi, aC[p]);
      aD[p] = wmma_bf16(aUi, bWr, aD[p]);
    }
  }
#pragma unroll
  for (int p = 0; p < 2; ++p) {
    v8f gr = aA[p] - aB[p];
    v8f gi = aC[p] + aD[p];
    int nt = ntb + p;
#pragma unroll
    for (int v = 0; v < 8; ++v) {
      int brow = mt*16 + v + 8*hi;
      int o = nt*16 + n;
      size_t base = (((size_t)brow*NH + h)*NO + o) * 128;
      gc[base + xm]      = (__bf16)gr[v];
      gc[base + 64 + xm] = (__bf16)gi[v];
    }
  }
}

// ---------------- K4: iDFT  out[o,t] = Gc[o,:] . basisI[:,t] ----------------
// grid 256 (b,h); writes d_out [B][H][O][L] fp32
__global__ __launch_bounds__(256) void k_idft(const __bf16* gc, const __bf16* basisI,
                                              float* out) {
  __shared__ __align__(16) __bf16 sG[16*512];    // A tiles [ks*4+mt]
  __shared__ __align__(16) __bf16 sBI[32*512];   // B tiles [ks*8+nt]
  const int bh = blockIdx.x;
  const int tid = threadIdx.x;
  const int w = tid >> 5, lane = tid & 31;
  const int mt = w >> 1, ntb = (w & 1) * 4;
  const int hi = lane >> 4, n = lane & 15;

  const __bf16* gb = gc + (size_t)bh*NO*128;
  float* ob = out + (size_t)bh*NO*LSEQ;
#pragma unroll 2
  for (int i = tid; i < 64*16; i += 256) {       // G as A: m=o, k=c
    int o = i >> 4, c0 = (i & 15) << 3;
    v8bf val = *(const v8bf*)(gb + (size_t)o*128 + c0);
    *(v8bf*)(sG + (((c0 >> 5) << 2) + (o >> 4))*512 + swzA(o & 15, c0 & 31)) = val;
  }
  for (int tc = 0; tc < 8; ++tc) {
#pragma unroll 4
    for (int i = tid; i < 128*16; i += 256) {    // basisI as B: k=c, n=t
      int tl = i >> 4, c0 = (i & 15) << 3;
      v8bf val = *(const v8bf*)(basisI + (size_t)(tc*128 + tl)*128 + c0);
      *(v8bf*)(sBI + (((c0 >> 5) << 3) + (tl >> 4))*512 + swzB(c0 & 31, tl & 15)) = val;
    }
    __syncthreads();
    v8f acc[4];
#pragma unroll
    for (int j = 0; j < 4; ++j) acc[j] = zero8();
#pragma unroll
    for (int ks = 0; ks < 4; ++ks) {
      v16bf a = load_frag(&sG[((ks << 2) + mt)*512]);
#pragma unroll
      for (int j = 0; j < 4; ++j) {
        v16bf bb = load_frag(&sBI[((ks << 3) + ntb + j)*512]);
        acc[j] = wmma_bf16(a, bb, acc[j]);
      }
    }
#pragma unroll
    for (int j = 0; j < 4; ++j) {
      int t = tc*128 + (ntb + j)*16 + n;
#pragma unroll
      for (int v = 0; v < 8; ++v) {
        int o = mt*16 + v + 8*hi;
        ob[(size_t)o*LSEQ + t] = acc[j][v];
      }
    }
    __syncthreads();
  }
}

// ---------------- launch ----------------
extern "C" void kernel_launch(void* const* d_in, const int* in_sizes, int n_in,
                              void* d_out, int out_size, void* d_ws, size_t ws_size,
                              hipStream_t stream) {
  const float* q  = (const float*)d_in[0];
  const float* k  = (const float*)d_in[1];
  // d_in[2] = v (unused by the reference computation)
  const float* Wr = (const float*)d_in[3];
  const float* Wi = (const float*)d_in[4];
  float* out = (float*)d_out;

  char* ws = (char*)d_ws;
  __bf16* basisF = (__bf16*)(ws + 0);          //  256 KB [c][t]
  __bf16* basisI = (__bf16*)(ws + 262144);     //  256 KB [t][c]
  __bf16* qft    = (__bf16*)(ws + 524288);     //    4 MB [bh][ri][x][e]
  __bf16* kf     = (__bf16*)(ws + 4718592);    //    4 MB [bh][ri][e][y]
  __bf16* kft    = (__bf16*)(ws + 8912896);    //    4 MB [bh][ri][y][e]
  __bf16* ux     = (__bf16*)(ws + 13107200);   //    4 MB [h][x][ri][b][e]
  __bf16* wb     = (__bf16*)(ws + 17301504);   //    8 MB [ri][h][x][o][e]
  __bf16* gc     = (__bf16*)(ws + 25690112);   //    4 MB [b][h][o][c]

  k_basis<<<256, 256, 0, stream>>>(basisF, basisI);
  k_prepw<<<1024, 256, 0, stream>>>(Wr, Wi, wb);
  k_dft  <<<512, 256, 0, stream>>>(q, k, basisF, qft, kf, kft);
  k_attn <<<256, 256, 0, stream>>>(qft, kf, kft, ux);
  k_wcon <<<512, 128, 0, stream>>>(ux, wb, gc);
  k_idft <<<256, 256, 0, stream>>>(gc, basisI, out);
}
